// GCN_11269994185058
// MI455X (gfx1250) — matrix operand order repaired and checked by reading, not compiled, not run on hardware
//
#include <hip/hip_runtime.h>
#include <hip/hip_fp16.h>

typedef __attribute__((ext_vector_type(16))) _Float16 v16h;
typedef __attribute__((ext_vector_type(8)))  float    v8f;

#define WAVES_PER_BLOCK 8
#define ROWS_PER_BLOCK (WAVES_PER_BLOCK * 16)
#define DD 64  // embedding dim

// ---------------------------------------------------------------------------
// GEMM: out[n x 64] = act(H)[n x K] @ W[K x 64]   (act = tanh if TANH)
// One wave computes a 16-row x 64-col tile via 8x v_wmma_f32_16x16x32_f16.
// W^T is staged in LDS so each B fragment is 2 contiguous ds_load_b128.
// ---------------------------------------------------------------------------
template<bool FULLK, bool TANH>
__global__ void __launch_bounds__(256)
gcn_gemm_wmma(const float* __restrict__ H, const float* __restrict__ W,
              float* __restrict__ out, int n, int K)
{
    // W transposed: WT[ncol][k], f16, zero-padded rows k >= K. 8 KB.
    __shared__ _Float16 WT[DD * DD];
    const int tid = threadIdx.x;
    for (int idx = tid; idx < DD * DD; idx += blockDim.x) {
        int ncol = idx >> 6, k = idx & 63;
        WT[idx] = (k < K) ? (_Float16)W[k * DD + ncol] : (_Float16)0.0f;
    }
    __syncthreads();

    const int wave    = tid >> 5;
    const int lane    = tid & 31;
    const int m       = lane & 15;
    const bool hiHalf = lane >= 16;
    const int base    = blockIdx.x * ROWS_PER_BLOCK + wave * 16;
    const int arow    = base + m;

    v8f acc[4] = {};

    constexpr int KSTEPS = FULLK ? 2 : 1;
#pragma unroll
    for (int ks = 0; ks < KSTEPS; ++ks) {
        const int k0 = ks << 5;

        // ---- A fragment: 16x32 f16 (lane<16: K k0+0..15;
        //                              lane>=16: +16) ----
        v16h a;
        const int kbase = k0 + (hiHalf ? 16 : 0);
        if (arow < n) {
            const float* hrow = H + (long)arow * K;
            if (FULLK) {
                // one aligned 64B run: [kbase, kbase+16)
                float4 u0 = *(const float4*)(hrow + kbase);
                float4 u1 = *(const float4*)(hrow + kbase + 4);
                float4 w0 = *(const float4*)(hrow + kbase + 8);
                float4 w1 = *(const float4*)(hrow + kbase + 12);
                float tmp[16] = {u0.x, u0.y, u0.z, u0.w, u1.x, u1.y, u1.z, u1.w,
                                 w0.x, w0.y, w0.z, w0.w, w1.x, w1.y, w1.z, w1.w};
#pragma unroll
                for (int e = 0; e < 16; ++e) {
                    float v = tmp[e];
                    if (TANH) v = tanhf(v);
                    a[e] = (_Float16)v;
                }
            } else {
#pragma unroll
                for (int e = 0; e < 16; ++e) {
                    int k = kbase + e;
                    float v = (k < K) ? hrow[k] : 0.0f;
                    if (TANH) v = tanhf(v);
                    a[e] = (_Float16)v;
                }
            }
        } else {
#pragma unroll
            for (int e = 0; e < 16; ++e) a[e] = (_Float16)0.0f;
        }

        // ---- B fragments: contiguous 32B per lane from transposed LDS ----
        const int kb = k0 + (hiHalf ? 16 : 0);
#pragma unroll
        for (int nt = 0; nt < 4; ++nt) {
            const int ncol = nt * 16 + m;
            v16h b = *(const v16h*)(&WT[ncol * DD + kb]);
            acc[nt] = __builtin_amdgcn_wmma_f32_16x16x32_f16(
                false, a, false, b, (short)0, acc[nt], false, false);
        }
    }

    // ---- store D: lane holds col = m, rows v + 8*hiHalf ----
#pragma unroll
    for (int nt = 0; nt < 4; ++nt) {
#pragma unroll
        for (int v = 0; v < 8; ++v) {
            int r = base + v + (hiHalf ? 8 : 0);
            if (r < n) out[(long)r * DD + nt * 16 + m] = acc[nt][v];
        }
    }
}

// ---------------------------------------------------------------------------
// Degree / normalization
// ---------------------------------------------------------------------------
__global__ void deg_init(float* __restrict__ deg, int n) {
    int i = blockIdx.x * blockDim.x + threadIdx.x;
    if (i < n) deg[i] = 1.0f;                  // self-loop
}
__global__ void deg_edges(const int* __restrict__ eidx, float* __restrict__ deg, int E) {
    int e = blockIdx.x * blockDim.x + threadIdx.x;
    if (e < E) atomicAdd(&deg[eidx[E + e]], 1.0f);   // col = edge_index[1]
}
__global__ void deg_rsqrt(float* __restrict__ deg, int n) {
    int i = blockIdx.x * blockDim.x + threadIdx.x;
    if (i < n) deg[i] = rsqrtf(deg[i]);        // in-place -> dinv
}

// out[i,d] = b[d] + dinv[i]^2 * hW[i,d]   (bias + self-loop seed)
__global__ void self_bias(const float* __restrict__ hW, const float* __restrict__ dinv,
                          const float* __restrict__ b, float* __restrict__ out, int n)
{
    int t = blockIdx.x * blockDim.x + threadIdx.x;
    if (t >= n * DD) return;
    int i = t >> 6, d = t & 63;
    float di = dinv[i];
    out[t] = b[d] + di * di * hW[t];
}

// 16 threads per edge, 4 floats each: out[col] += dinv[row]*dinv[col]*hW[row]
__global__ void scatter_edges(const int* __restrict__ eidx, const float* __restrict__ dinv,
                              const float* __restrict__ hW, float* __restrict__ out, int E)
{
    long t = (long)blockIdx.x * blockDim.x + threadIdx.x;
    int e = (int)(t >> 4);
    if (e >= E) return;
    int sub = (int)(t & 15);
    int r = eidx[e];
    int c = eidx[E + e];
    float norm = dinv[r] * dinv[c];
    const float4 s = *(const float4*)(hW + (long)r * DD + sub * 4);
    float* dst = out + (long)c * DD + sub * 4;
    atomicAdd(dst + 0, norm * s.x);
    atomicAdd(dst + 1, norm * s.y);
    atomicAdd(dst + 2, norm * s.z);
    atomicAdd(dst + 3, norm * s.w);
}

// ---------------------------------------------------------------------------
// Pooling: segment max / mean over sorted batch ids, tanh fused on load
// ---------------------------------------------------------------------------
__device__ __forceinline__ unsigned fkey(float f) {
    int i = __float_as_int(f);
    return (unsigned)(i ^ ((i >> 31) | 0x80000000));   // monotone float->uint
}
__device__ __forceinline__ float funkey(unsigned u) {
    int i = (u & 0x80000000u) ? (int)(u ^ 0x80000000u) : ~(int)u;
    return __int_as_float(i);
}

__global__ void pool_init(unsigned* __restrict__ gmax, float* __restrict__ gsum,
                          float* __restrict__ cnt, int G)
{
    int t = blockIdx.x * blockDim.x + threadIdx.x;
    if (t < G * DD) { gmax[t] = fkey(-INFINITY); gsum[t] = 0.0f; }
    if (t < G) cnt[t] = 0.0f;
}

__global__ void pool_scatter(const float* __restrict__ h, const int* __restrict__ batch,
                             unsigned* __restrict__ gmax, float* __restrict__ gsum,
                             float* __restrict__ cnt, int n)
{
    int t = blockIdx.x * blockDim.x + threadIdx.x;
    if (t >= n * DD) return;
    int i = t >> 6, d = t & 63;
    int g = batch[i];
    float v = tanhf(h[t]);
    atomicMax(&gmax[g * DD + d], fkey(v));
    atomicAdd(&gsum[g * DD + d], v);
    if (d == 0) atomicAdd(&cnt[g], 1.0f);
}

__global__ void pool_out(const unsigned* __restrict__ gmax, const float* __restrict__ gsum,
                         const float* __restrict__ cnt, const float* __restrict__ Wout,
                         const float* __restrict__ bout, float* __restrict__ out, int G)
{
    int g = blockIdx.x * blockDim.x + threadIdx.x;
    if (g >= G) return;
    float inv = 1.0f / fmaxf(cnt[g], 1.0f);
    float acc = bout[0];
#pragma unroll 4
    for (int d = 0; d < DD; ++d) {
        acc += funkey(gmax[g * DD + d]) * Wout[d];
        acc += (gsum[g * DD + d] * inv) * Wout[DD + d];
    }
    out[g] = acc;
}

// ---------------------------------------------------------------------------
extern "C" void kernel_launch(void* const* d_in, const int* in_sizes, int n_in,
                              void* d_out, int out_size, void* d_ws, size_t ws_size,
                              hipStream_t stream)
{
    const float* x     = (const float*)d_in[0];
    const int*   eidx  = (const int*)  d_in[1];
    const int*   batch = (const int*)  d_in[2];
    const float* W0 = (const float*)d_in[3];  const float* b0 = (const float*)d_in[4];
    const float* W1 = (const float*)d_in[5];  const float* b1 = (const float*)d_in[6];
    const float* W2 = (const float*)d_in[7];  const float* b2 = (const float*)d_in[8];
    const float* W3 = (const float*)d_in[9];  const float* b3 = (const float*)d_in[10];
    const float* Wout = (const float*)d_in[11];
    const float* bout = (const float*)d_in[12];
    float* out = (float*)d_out;

    const int N   = in_sizes[2];
    const int E   = in_sizes[1] / 2;
    const int DIN = in_sizes[0] / N;
    const int G   = out_size;

    // workspace carve-out (256B aligned)
    size_t off = 0;
    char* ws = (char*)d_ws;
    auto carve = [&](size_t bytes) -> void* {
        void* p = ws + off;
        off = (off + bytes + 255) & ~(size_t)255;
        return p;
    };
    float*    dinv  = (float*)   carve((size_t)N * sizeof(float));
    float*    bufA  = (float*)   carve((size_t)N * DD * sizeof(float));
    float*    bufB  = (float*)   carve((size_t)N * DD * sizeof(float));  // hW scratch
    float*    bufC  = (float*)   carve((size_t)N * DD * sizeof(float));
    unsigned* gmax  = (unsigned*)carve((size_t)G * DD * sizeof(unsigned));
    float*    gsum  = (float*)   carve((size_t)G * DD * sizeof(float));
    float*    cnt   = (float*)   carve((size_t)G * sizeof(float));
    (void)ws_size;

    const int T = 256;
    const int gemmBlocks = (N + ROWS_PER_BLOCK - 1) / ROWS_PER_BLOCK;
    const int ndBlocks   = (N * DD + T - 1) / T;
    const int edgeBlocks = (int)(((long)E * 16 + T - 1) / T);

    // --- normalization: dinv = rsqrt(1 + in-degree) ---
    deg_init <<<(N + T - 1) / T, T, 0, stream>>>(dinv, N);
    deg_edges<<<(E + T - 1) / T, T, 0, stream>>>(eidx, dinv, E);
    deg_rsqrt<<<(N + T - 1) / T, T, 0, stream>>>(dinv, N);

    // one GCN conv layer: dst = b + D^-1/2 (A+I) D^-1/2 (act(src) @ W)
    auto conv = [&](const float* src, int K, bool tanh_,
                    const float* W, const float* b, float* dst) {
        if (K == DD) {
            if (tanh_) gcn_gemm_wmma<true,  true ><<<gemmBlocks, T, 0, stream>>>(src, W, bufB, N, K);
            else       gcn_gemm_wmma<true,  false><<<gemmBlocks, T, 0, stream>>>(src, W, bufB, N, K);
        } else {
            if (tanh_) gcn_gemm_wmma<false, true ><<<gemmBlocks, T, 0, stream>>>(src, W, bufB, N, K);
            else       gcn_gemm_wmma<false, false><<<gemmBlocks, T, 0, stream>>>(src, W, bufB, N, K);
        }
        self_bias    <<<ndBlocks,   T, 0, stream>>>(bufB, dinv, b, dst, N);
        scatter_edges<<<edgeBlocks, T, 0, stream>>>(eidx, dinv, bufB, dst, E);
    };

    conv(x,    DIN, false, W0, b0, bufA);   // h0 (pre-tanh) in bufA
    conv(bufA, DD,  true,  W1, b1, bufC);   // tanh fused into GEMM A-load
    conv(bufC, DD,  true,  W2, b2, bufA);
    conv(bufA, DD,  true,  W3, b3, bufC);   // final pre-tanh h in bufC

    // --- pooling + readout (tanh fused) ---
    pool_init   <<<(G * DD + T - 1) / T, T, 0, stream>>>(gmax, gsum, cnt, G);
    pool_scatter<<<ndBlocks, T, 0, stream>>>(bufC, batch, gmax, gsum, cnt, N);
    pool_out    <<<(G + T - 1) / T, T, 0, stream>>>(gmax, gsum, cnt, Wout, bout, out, G);
}
